// GatedAttention_6665789243993
// MI455X (gfx1250) — matrix operand order repaired and checked by reading, not compiled, not run on hardware
//
#include <hip/hip_runtime.h>
#include <hip/hip_bf16.h>
#include <math.h>

typedef __bf16 bf16;
typedef __attribute__((ext_vector_type(16))) __bf16 v16bf;
typedef __attribute__((ext_vector_type(8)))  __bf16 v8bf;
typedef __attribute__((ext_vector_type(8)))  float  v8f;

#define NB   8
#define SLD  2048
#define SLM  512
#define DIN  1024
#define HID  1024

// ---------------------------------------------------------------------------
// LDS tile convention: tiles are [rows][32] bf16 (64B pitch).
//   A tiles: row = M index, 32 K values per row.
//   B tiles: row = N index, 32 K values per row (i.e. stored K-transposed).
// ---------------------------------------------------------------------------

// A fragment (16x32 bf16, V_WMMA_F32_16X16X32_BF16 layout):
// lanes 0-15 : row M=lane,   elems 0-7 = K 0..7,  elems 8-15 = K 16..23
// lanes 16-31: row M=lane-16,elems 0-7 = K 8..15, elems 8-15 = K 24..31
static __device__ __forceinline__ v16bf frag_a(const bf16* tile, int rowBase, int lane) {
  const int r  = rowBase + (lane & 15);
  const int kb = (lane >> 4) << 3;
  const bf16* p = tile + r * 32 + kb;
  v8bf lo = *(const v8bf*)(p);
  v8bf hi = *(const v8bf*)(p + 16);
  return __builtin_shufflevector(lo, hi, 0,1,2,3,4,5,6,7,8,9,10,11,12,13,14,15);
}

// B fragment (32x16 bf16): lane n (0-15) holds K=0..7,16..23, lane n+16 holds K=8..15,24..31.
// With the [n][k] LDS layout this is two 16B reads.
static __device__ __forceinline__ v16bf frag_b(const bf16* tile, int nBase, int lane) {
  const int n  = nBase + (lane & 15);
  const int kh = (lane >> 4) << 3;
  const bf16* p = tile + n * 32 + kh;
  v8bf lo = *(const v8bf*)(p);
  v8bf hi = *(const v8bf*)(p + 16);
  return __builtin_shufflevector(lo, hi, 0,1,2,3,4,5,6,7,8,9,10,11,12,13,14,15);
}

static __device__ __forceinline__ v8f wmma_bf16(v16bf a, v16bf b, v8f c) {
  return __builtin_amdgcn_wmma_f32_16x16x32_bf16(false, a, false, b, (short)0, c, false, false);
}

// ---------------------------------------------------------------------------
// Kernel 1: C[m][n] = relu(A[m][:K] @ W[:K][n] + bias[n])   (f32 in, bf16 out)
// ---------------------------------------------------------------------------
__global__ __launch_bounds__(256) void k_proj_relu(
    const float* __restrict__ A, const float* __restrict__ W,
    const float* __restrict__ bias, bf16* __restrict__ C,
    int K, int Nn)
{
  __shared__ __align__(32) bf16 As[64 * 32];
  __shared__ __align__(32) bf16 Bs[128 * 32];
  const int tid  = threadIdx.x;
  const int lane = tid & 31;
  const int wid  = tid >> 5;
  const int row0 = blockIdx.x * 64;
  const int col0 = blockIdx.y * 128;
  const int wr   = (wid >> 2) * 32;
  const int wc   = (wid & 3)  * 32;

  v8f acc[2][2] = {};

  for (int k0 = 0; k0 < K; k0 += 32) {
    if (k0) __syncthreads();
    { // A tile 64x32: f32 -> bf16
      const int r  = tid >> 2;
      const int ks = (tid & 3) * 8;
      const float* g = A + (size_t)(row0 + r) * K + (k0 + ks);
      bf16* d = As + r * 32 + ks;
#pragma unroll
      for (int i = 0; i < 8; ++i) d[i] = (bf16)g[i];
    }
    { // B tile: W[k][n] -> Bs[n][k]  (transpose while converting)
      const int kk = tid >> 3;
      const int ns = (tid & 7) * 16;
      const float* g = W + (size_t)(k0 + kk) * Nn + (col0 + ns);
      if (k0 + 32 < K) __builtin_prefetch(g + (size_t)32 * Nn, 0, 0);
#pragma unroll
      for (int i = 0; i < 16; ++i) Bs[(ns + i) * 32 + kk] = (bf16)g[i];
    }
    __syncthreads();
    const v16bf a0 = frag_a(As, wr,      lane);
    const v16bf a1 = frag_a(As, wr + 16, lane);
    const v16bf b0 = frag_b(Bs, wc,      lane);
    const v16bf b1 = frag_b(Bs, wc + 16, lane);
    acc[0][0] = wmma_bf16(a0, b0, acc[0][0]);
    acc[0][1] = wmma_bf16(a0, b1, acc[0][1]);
    acc[1][0] = wmma_bf16(a1, b0, acc[1][0]);
    acc[1][1] = wmma_bf16(a1, b1, acc[1][1]);
  }

  const int nlo = lane & 15;
  const int mhi = (lane >> 4) << 3;
#pragma unroll
  for (int i = 0; i < 2; ++i)
#pragma unroll
    for (int j = 0; j < 2; ++j) {
      const int n  = col0 + wc + 16 * j + nlo;
      const float bv = bias[n];
      const int mb = row0 + wr + 16 * i + mhi;
#pragma unroll
      for (int r = 0; r < 8; ++r) {
        const float v = acc[i][j][r] + bv;
        C[(size_t)(mb + r) * Nn + n] = (bf16)(v > 0.f ? v : 0.f);
      }
    }
}

// ---------------------------------------------------------------------------
// Kernel 2: scores[b][l][m] = (idot[b][l][:] . mdot[b][m][:]) / 32
// B = mdot^T, but mdot rows are exactly the [n][k] LDS layout -> direct copy.
// ---------------------------------------------------------------------------
__global__ __launch_bounds__(256) void k_scores(
    const bf16* __restrict__ idot, const bf16* __restrict__ mdot,
    float* __restrict__ S)
{
  __shared__ __align__(32) bf16 As[64 * 32];
  __shared__ __align__(32) bf16 Bs[128 * 32];
  const int tid  = threadIdx.x;
  const int lane = tid & 31;
  const int wid  = tid >> 5;
  const int b    = blockIdx.z;
  const bf16* A  = idot + (size_t)b * SLD * HID;
  const bf16* B  = mdot + (size_t)b * SLM * HID;
  float* out     = S    + (size_t)b * SLD * SLM;
  const int row0 = blockIdx.x * 64;
  const int col0 = blockIdx.y * 128;
  const int wr   = (wid >> 2) * 32;
  const int wc   = (wid & 3)  * 32;

  v8f acc[2][2] = {};

  for (int k0 = 0; k0 < HID; k0 += 32) {
    if (k0) __syncthreads();
    { // A tile 64x32 bf16 copy
      const int r  = tid >> 2;
      const int ks = (tid & 3) * 8;
      *(v8bf*)(As + r * 32 + ks) =
          *(const v8bf*)(A + (size_t)(row0 + r) * HID + (k0 + ks));
    }
    { // B tile: rows n of mdot, 16 bf16 per thread
      const int n  = tid >> 1;
      const int ks = (tid & 1) * 16;
      const bf16* g = B + (size_t)(col0 + n) * HID + (k0 + ks);
      *(v8bf*)(Bs + n * 32 + ks)     = *(const v8bf*)(g);
      *(v8bf*)(Bs + n * 32 + ks + 8) = *(const v8bf*)(g + 8);
    }
    __syncthreads();
    const v16bf a0 = frag_a(As, wr,      lane);
    const v16bf a1 = frag_a(As, wr + 16, lane);
    const v16bf b0 = frag_b(Bs, wc,      lane);
    const v16bf b1 = frag_b(Bs, wc + 16, lane);
    acc[0][0] = wmma_bf16(a0, b0, acc[0][0]);
    acc[0][1] = wmma_bf16(a0, b1, acc[0][1]);
    acc[1][0] = wmma_bf16(a1, b0, acc[1][0]);
    acc[1][1] = wmma_bf16(a1, b1, acc[1][1]);
  }

  const int nlo = lane & 15;
  const int mhi = (lane >> 4) << 3;
#pragma unroll
  for (int i = 0; i < 2; ++i)
#pragma unroll
    for (int j = 0; j < 2; ++j) {
      const int n  = col0 + wc + 16 * j + nlo;
      const int mb = row0 + wr + 16 * i + mhi;
#pragma unroll
      for (int r = 0; r < 8; ++r)
        out[(size_t)(mb + r) * SLM + n] = acc[i][j][r] * 0.03125f; // 1/sqrt(1024)
    }
}

// ---------------------------------------------------------------------------
// Kernel 3: masked softmax over the 512-wide memory axis. One wave per row.
// ---------------------------------------------------------------------------
__global__ __launch_bounds__(256) void k_softmax(
    const float* __restrict__ S, const float* __restrict__ mask,
    bf16* __restrict__ Wt)
{
  const int row  = blockIdx.x * 8 + (threadIdx.x >> 5);
  const int lane = threadIdx.x & 31;
  const int b    = row >> 11;                 // row / SLD
  const float* s  = S    + (size_t)row * SLM;
  const float* mk = mask + (size_t)b   * SLM;
  float v[16];
  float mx = -3.0e38f;
#pragma unroll
  for (int j = 0; j < 16; ++j) {
    const int c = lane + j * 32;
    const float t = s[c] - 1.0e30f * (1.0f - mk[c]);
    v[j] = t;
    mx = fmaxf(mx, t);
  }
  for (int off = 16; off > 0; off >>= 1) mx = fmaxf(mx, __shfl_xor(mx, off, 32));
  float sum = 0.f;
#pragma unroll
  for (int j = 0; j < 16; ++j) { v[j] = __expf(v[j] - mx); sum += v[j]; }
  for (int off = 16; off > 0; off >>= 1) sum += __shfl_xor(sum, off, 32);
  const float inv = 1.0f / sum;
  bf16* w = Wt + (size_t)row * SLM;
#pragma unroll
  for (int j = 0; j < 16; ++j) w[lane + j * 32] = (bf16)(v[j] * inv);
}

// ---------------------------------------------------------------------------
// Kernel 4: out1[b][l][d] = weights[b][l][:512] @ mem[b][:512][d]   (bf16 out)
// ---------------------------------------------------------------------------
__global__ __launch_bounds__(256) void k_out1(
    const bf16* __restrict__ Wt, const float* __restrict__ mem,
    bf16* __restrict__ O1)
{
  __shared__ __align__(32) bf16 As[64 * 32];
  __shared__ __align__(32) bf16 Bs[128 * 32];
  const int tid  = threadIdx.x;
  const int lane = tid & 31;
  const int wid  = tid >> 5;
  const int b    = blockIdx.z;
  const bf16*  A  = Wt  + (size_t)b * SLD * SLM;
  const float* Bm = mem + (size_t)b * SLM * DIN;
  bf16* out       = O1  + (size_t)b * SLD * DIN;
  const int row0 = blockIdx.x * 64;
  const int col0 = blockIdx.y * 128;
  const int wr   = (wid >> 2) * 32;
  const int wc   = (wid & 3)  * 32;

  v8f acc[2][2] = {};

  for (int k0 = 0; k0 < SLM; k0 += 32) {
    if (k0) __syncthreads();
    { // A tile bf16 copy (lda = 512)
      const int r  = tid >> 2;
      const int ks = (tid & 3) * 8;
      *(v8bf*)(As + r * 32 + ks) =
          *(const v8bf*)(A + (size_t)(row0 + r) * SLM + (k0 + ks));
    }
    { // B tile: mem[k][d] -> Bs[n][k], f32 -> bf16
      const int kk = tid >> 3;
      const int ns = (tid & 7) * 16;
      const float* g = Bm + (size_t)(k0 + kk) * DIN + (col0 + ns);
      if (k0 + 32 < SLM) __builtin_prefetch(g + (size_t)32 * DIN, 0, 0);
#pragma unroll
      for (int i = 0; i < 16; ++i) Bs[(ns + i) * 32 + kk] = (bf16)g[i];
    }
    __syncthreads();
    const v16bf a0 = frag_a(As, wr,      lane);
    const v16bf a1 = frag_a(As, wr + 16, lane);
    const v16bf b0 = frag_b(Bs, wc,      lane);
    const v16bf b1 = frag_b(Bs, wc + 16, lane);
    acc[0][0] = wmma_bf16(a0, b0, acc[0][0]);
    acc[0][1] = wmma_bf16(a0, b1, acc[0][1]);
    acc[1][0] = wmma_bf16(a1, b0, acc[1][0]);
    acc[1][1] = wmma_bf16(a1, b1, acc[1][1]);
  }

  const int nlo = lane & 15;
  const int mhi = (lane >> 4) << 3;
#pragma unroll
  for (int i = 0; i < 2; ++i)
#pragma unroll
    for (int j = 0; j < 2; ++j) {
      const int n  = col0 + wc + 16 * j + nlo;
      const int mb = row0 + wr + 16 * i + mhi;
#pragma unroll
      for (int r = 0; r < 8; ++r)
        out[(size_t)(mb + r) * DIN + n] = (bf16)acc[i][j][r];
    }
}

// ---------------------------------------------------------------------------
// Kernel 5: pre = [x | out1] @ W2 + b2 ; out = sigmoid(pre) * tanh(pre)
// K = 2048 split as k<1024 from x (f32), k>=1024 from out1 (bf16).
// ---------------------------------------------------------------------------
__global__ __launch_bounds__(256) void k_gate(
    const float* __restrict__ X, const bf16* __restrict__ O1,
    const float* __restrict__ W2, const float* __restrict__ b2,
    float* __restrict__ Out)
{
  __shared__ __align__(32) bf16 As[64 * 32];
  __shared__ __align__(32) bf16 Bs[128 * 32];
  const int tid  = threadIdx.x;
  const int lane = tid & 31;
  const int wid  = tid >> 5;
  const int row0 = blockIdx.x * 64;
  const int col0 = blockIdx.y * 128;
  const int wr   = (wid >> 2) * 32;
  const int wc   = (wid & 3)  * 32;

  v8f acc[2][2] = {};

  for (int k0 = 0; k0 < 2 * DIN; k0 += 32) {
    if (k0) __syncthreads();
    { // A tile: first half from x (f32), second half from out1 (bf16)
      const int r  = tid >> 2;
      const int ks = (tid & 3) * 8;
      bf16* d = As + r * 32 + ks;
      if (k0 < DIN) {
        const float* g = X + (size_t)(row0 + r) * DIN + (k0 + ks);
#pragma unroll
        for (int i = 0; i < 8; ++i) d[i] = (bf16)g[i];
      } else {
        *(v8bf*)d = *(const v8bf*)(O1 + (size_t)(row0 + r) * DIN + (k0 - DIN + ks));
      }
    }
    { // B tile: W2[k][n] -> Bs[n][k]
      const int kk = tid >> 3;
      const int ns = (tid & 7) * 16;
      const float* g = W2 + (size_t)(k0 + kk) * HID + (col0 + ns);
      if (k0 + 32 < 2 * DIN) __builtin_prefetch(g + (size_t)32 * HID, 0, 0);
#pragma unroll
      for (int i = 0; i < 16; ++i) Bs[(ns + i) * 32 + kk] = (bf16)g[i];
    }
    __syncthreads();
    const v16bf a0 = frag_a(As, wr,      lane);
    const v16bf a1 = frag_a(As, wr + 16, lane);
    const v16bf b0 = frag_b(Bs, wc,      lane);
    const v16bf b1 = frag_b(Bs, wc + 16, lane);
    acc[0][0] = wmma_bf16(a0, b0, acc[0][0]);
    acc[0][1] = wmma_bf16(a0, b1, acc[0][1]);
    acc[1][0] = wmma_bf16(a1, b0, acc[1][0]);
    acc[1][1] = wmma_bf16(a1, b1, acc[1][1]);
  }

  const int nlo = lane & 15;
  const int mhi = (lane >> 4) << 3;
#pragma unroll
  for (int i = 0; i < 2; ++i)
#pragma unroll
    for (int j = 0; j < 2; ++j) {
      const int n  = col0 + wc + 16 * j + nlo;
      const float bv = b2[n];
      const int mb = row0 + wr + 16 * i + mhi;
#pragma unroll
      for (int r = 0; r < 8; ++r) {
        const float p = acc[i][j][r] + bv;
        const float sg = 1.0f / (1.0f + __expf(-p));
        Out[(size_t)(mb + r) * HID + n] = sg * tanhf(p);
      }
    }
}

// ---------------------------------------------------------------------------
extern "C" void kernel_launch(void* const* d_in, const int* in_sizes, int n_in,
                              void* d_out, int out_size, void* d_ws, size_t ws_size,
                              hipStream_t stream) {
  (void)in_sizes; (void)n_in; (void)out_size; (void)ws_size;
  const float* x   = (const float*)d_in[0];  // [8,2048,1024]
  const float* mem = (const float*)d_in[1];  // [8,512,1024]
  const float* msk = (const float*)d_in[2];  // [8,512]
  const float* W1  = (const float*)d_in[3];  // [1024,1024]
  const float* b1  = (const float*)d_in[4];
  const float* Wm  = (const float*)d_in[5];  // [1024,1024]
  const float* bm  = (const float*)d_in[6];
  const float* W2  = (const float*)d_in[7];  // [2048,1024]
  const float* b2  = (const float*)d_in[8];
  float* out = (float*)d_out;                // [8,2048,1024] fp32 ++ [8,512,1024] fp32

  // Workspace layout (bytes)
  char* ws = (char*)d_ws;
  bf16*  mdot   = (bf16*)(ws);                                    //  8 MB
  bf16*  idot   = (bf16*)(ws + (size_t)8388608);                  // 32 MB
  float* scores = (float*)(ws + (size_t)41943040);                // 32 MB
  bf16*  wghts  = (bf16*)(ws + (size_t)75497472);                 // 16 MB
  bf16*  out1   = (bf16*)(ws + (size_t)92274688);                 // 32 MB

  const dim3 blk(256);
  // memory_dot = relu(mem @ Wm + bm)   M = 8*512 flattened
  k_proj_relu<<<dim3((NB * SLM) / 64, HID / 128), blk, 0, stream>>>(mem, Wm, bm, mdot, DIN, HID);
  // input_dot = relu(x @ W1 + b1)      M = 8*2048 flattened
  k_proj_relu<<<dim3((NB * SLD) / 64, HID / 128), blk, 0, stream>>>(x, W1, b1, idot, DIN, HID);
  // scores = idot @ mdot^T / 32        per batch
  k_scores<<<dim3(SLD / 64, SLM / 128, NB), blk, 0, stream>>>(idot, mdot, scores);
  // masked softmax over memory axis
  k_softmax<<<dim3((NB * SLD) / 8), blk, 0, stream>>>(scores, msk, wghts);
  // out1 = weights @ mem               per batch
  k_out1<<<dim3(SLD / 64, DIN / 128, NB), blk, 0, stream>>>(wghts, mem, out1);
  // gate: [x|out1] @ W2 + b2 -> sigmoid*tanh
  k_gate<<<dim3((NB * SLD) / 64, HID / 128), blk, 0, stream>>>(x, out1, W2, b2, out);
  // second tuple element: mem passthrough
  hipMemcpyAsync(out + (size_t)NB * SLD * HID, mem,
                 (size_t)NB * SLM * DIN * sizeof(float),
                 hipMemcpyDeviceToDevice, stream);
}
